// TransformerBlock_40845138985150
// MI455X (gfx1250) — compile-verified, matrix operand order checked by reading, not run on hardware
//
#include <hip/hip_runtime.h>
#include <hip/hip_bf16.h>

typedef __bf16 bf16;
typedef __attribute__((ext_vector_type(16))) __bf16 v16bf;
typedef __attribute__((ext_vector_type(8)))  float  v8f;
typedef __attribute__((ext_vector_type(4)))  int    v4i;

#define EMB   1024
#define HEADS 8
#define HID   (4*EMB)
#define BATCH 2
#define SEQ   2048
#define BT    (BATCH*SEQ)     // 4096
#define HD    (HEADS*EMB)     // 8192
#define LN_EPS 1e-5f

#define BM 128
#define BN 128
#define BK 32
#define GT 256                // threads per GEMM block (8 waves of 32)

// CDNA5 async copy engine (global -> LDS without VGPR round-trip), guarded so
// either toolchain compiles; fallback is global_load + ds_store.
#if defined(__gfx1250__) && __has_builtin(__builtin_amdgcn_global_load_async_to_lds_b128)
#define USE_ASYNC_LDS 1
typedef __attribute__((address_space(1))) v4i* gptr_b128;
typedef __attribute__((address_space(3))) v4i* lptr_b128;
#define ASYNC_B128(gp, lp) \
    __builtin_amdgcn_global_load_async_to_lds_b128( \
        (gptr_b128)(void*)(gp), (lptr_b128)(void*)(lp), 0, 0)
#else
#define USE_ASYNC_LDS 0
#endif

#if defined(__gfx1250__) && __has_builtin(__builtin_amdgcn_s_wait_asynccnt)
#define WAIT_ASYNC() __builtin_amdgcn_s_wait_asynccnt(0)
#else
#define WAIT_ASYNC() ((void)0)
#endif

// ---------------------------------------------------------------------------
// fp32 -> bf16 conversion (grid-stride)
// ---------------------------------------------------------------------------
__global__ __launch_bounds__(256)
void cvt_f32_bf16(const float* __restrict__ in, bf16* __restrict__ out, size_t n) {
    size_t i = (size_t)blockIdx.x * blockDim.x + threadIdx.x;
    size_t stride = (size_t)gridDim.x * blockDim.x;
    for (; i < n; i += stride) out[i] = (bf16)in[i];
}

// ---------------------------------------------------------------------------
// bf16 WMMA GEMM:  C[M,N] = alpha * A[M,K] (x) B  (+ bias) (ReLU optional)
//   TB=false: B is [K,N] row-major.  TB=true: B is [N,K] row-major (B^T).
// Batch base offsets:  ptr += (z/zdiv)*Outer + (z%zdiv)*Inner
// Block tile 128x128x32; 8 waves; wave tile 32x64 (2x4 WMMA 16x16 tiles).
// ---------------------------------------------------------------------------
template<bool TB, bool RELU, typename OutT>
__global__ __launch_bounds__(GT)
void gemm_wmma(const bf16* __restrict__ A, const bf16* __restrict__ Bm,
               OutT* __restrict__ C,
               int M, int N, int K, int lda, int ldb, int ldc,
               size_t aO, size_t aI, size_t bO, size_t bI, size_t cO, size_t cI,
               int zdiv, const float* __restrict__ bias, float alpha, int causal)
{
    const int m0 = blockIdx.y * BM;
    const int n0 = blockIdx.x * BN;
    if (causal && n0 > m0 + (BM - 1)) return;   // fully-masked causal block

    const int z = blockIdx.z;
    A  += (size_t)(z / zdiv) * aO + (size_t)(z % zdiv) * aI;
    Bm += (size_t)(z / zdiv) * bO + (size_t)(z % zdiv) * bI;
    C  += (size_t)(z / zdiv) * cO + (size_t)(z % zdiv) * cI;

    __shared__ __align__(16) unsigned short As[BM * BK];   // [m][k]
    __shared__ __align__(16) unsigned short Bs[BN * BK];   // [n][k]

    const int tid  = threadIdx.x;
    const int lane = tid & 31;
    const int wave = tid >> 5;
    const int wm   = wave >> 1;        // 0..3  (M direction)
    const int wn   = wave & 1;         // 0..1  (N direction)
    const int half = lane >> 4;        // lane group 0/1
    const int lrow = lane & 15;

    v8f acc[2][4];
#pragma unroll
    for (int mi = 0; mi < 2; ++mi)
#pragma unroll
        for (int ni = 0; ni < 4; ++ni)
            acc[mi][ni] = (v8f){0.f,0.f,0.f,0.f,0.f,0.f,0.f,0.f};

    for (int k0 = 0; k0 < K; k0 += BK) {
        // ---- stage A tile: 128x32 bf16, 2 x 16B per thread ----
#if USE_ASYNC_LDS
#pragma unroll
        for (int i = 0; i < 2; ++i) {
            int idx = tid + i * GT;              // 0..511 (each = one b128)
            int r   = idx >> 2;
            int kk  = (idx & 3) * 8;
            const bf16* gp = A + (size_t)(m0 + r) * lda + k0 + kk;
            ASYNC_B128(gp, As + idx * 8);
        }
#else
#pragma unroll
        for (int i = 0; i < 2; ++i) {
            int idx = tid + i * GT;
            int r   = idx >> 2;
            int kk  = (idx & 3) * 8;
            uint4 v = *(const uint4*)(A + (size_t)(m0 + r) * lda + k0 + kk);
            ((uint4*)As)[idx] = v;
        }
#endif
        // ---- stage B tile into [n][k] layout ----
        if constexpr (TB) {
            // B^T: B stored [N,K] row-major -> rows copy straight in
#if USE_ASYNC_LDS
#pragma unroll
            for (int i = 0; i < 2; ++i) {
                int idx = tid + i * GT;
                int r   = idx >> 2;
                int kk  = (idx & 3) * 8;
                const bf16* gp = Bm + (size_t)(n0 + r) * ldb + k0 + kk;
                ASYNC_B128(gp, Bs + idx * 8);
            }
#else
#pragma unroll
            for (int i = 0; i < 2; ++i) {
                int idx = tid + i * GT;
                int r   = idx >> 2;
                int kk  = (idx & 3) * 8;
                uint4 v = *(const uint4*)(Bm + (size_t)(n0 + r) * ldb + k0 + kk);
                ((uint4*)Bs)[idx] = v;
            }
#endif
        } else {
            // B stored [K,N] row-major -> transpose through LDS
            int kr  = tid >> 3;                  // 0..31
            int nc0 = (tid & 7) * 16;            // 0..112
            const bf16* src = Bm + (size_t)(k0 + kr) * ldb + n0 + nc0;
            unsigned short tmp[16];
            *(uint4*)&tmp[0] = *(const uint4*)(src);
            *(uint4*)&tmp[8] = *(const uint4*)(src + 8);
#pragma unroll
            for (int e = 0; e < 16; ++e)
                Bs[(size_t)(nc0 + e) * BK + kr] = tmp[e];
        }
        // prefetch next A tile into L2 while we compute
        if (k0 + BK < K)
            __builtin_prefetch(A + (size_t)(m0 + (tid >> 1)) * lda + k0 + BK, 0, 1);
        WAIT_ASYNC();
        __syncthreads();

        // ---- build fragments per ISA 16-bit A/B VGPR layout ----
        union Frag { v16bf v; unsigned int u[8]; };
        Frag af[2], bfr[4];
#pragma unroll
        for (int j = 0; j < 8; ++j) {
            // VGPR j holds bf16 pair at k = (j<4 ? 2j : 2j+8) + 8*half
            int ko = ((j < 4) ? 2 * j : 2 * j + 8) + 8 * half;
#pragma unroll
            for (int mi = 0; mi < 2; ++mi)
                af[mi].u[j] = *(const unsigned int*)
                    &As[(size_t)(wm * 32 + mi * 16 + lrow) * BK + ko];
#pragma unroll
            for (int ni = 0; ni < 4; ++ni)
                bfr[ni].u[j] = *(const unsigned int*)
                    &Bs[(size_t)(wn * 64 + ni * 16 + lrow) * BK + ko];
        }
#pragma unroll
        for (int mi = 0; mi < 2; ++mi)
#pragma unroll
            for (int ni = 0; ni < 4; ++ni)
                acc[mi][ni] = __builtin_amdgcn_wmma_f32_16x16x32_bf16(
                    false, af[mi].v, false, bfr[ni].v,
                    (short)0, acc[mi][ni], false, false);
        __syncthreads();
    }

    // ---- epilogue: alpha, bias, relu, store ----
#pragma unroll
    for (int mi = 0; mi < 2; ++mi) {
#pragma unroll
        for (int ni = 0; ni < 4; ++ni) {
            int col = n0 + wn * 64 + ni * 16 + lrow;
            float bv = bias ? bias[col] : 0.0f;
#pragma unroll
            for (int r = 0; r < 8; ++r) {
                int row = m0 + wm * 32 + mi * 16 + half * 8 + r;
                float v = alpha * acc[mi][ni][r] + bv;
                if (RELU) v = v > 0.f ? v : 0.f;
                C[(size_t)row * ldc + col] = (OutT)v;
            }
        }
    }
}

// ---------------------------------------------------------------------------
// Causal softmax over one row (b,h,t): reads fp32 scores, writes bf16 probs,
// zeros the masked tail so P@V can run as a dense GEMM.
// ---------------------------------------------------------------------------
__global__ __launch_bounds__(256)
void softmax_causal(const float* __restrict__ S, bf16* __restrict__ P, int T) {
    const int row = blockIdx.x;          // over B*H*T
    const int t   = row % T;
    const float* s = S + (size_t)row * T;
    bf16* p = P + (size_t)row * T;
    const int valid = t + 1;
    __shared__ float red[256];
    const int tid = threadIdx.x;

    float mx = -3.0e38f;
    for (int i = tid; i < valid; i += 256) mx = fmaxf(mx, s[i]);
    red[tid] = mx; __syncthreads();
    for (int w = 128; w > 0; w >>= 1) {
        if (tid < w) red[tid] = fmaxf(red[tid], red[tid + w]);
        __syncthreads();
    }
    mx = red[0]; __syncthreads();

    float sum = 0.f;
    for (int i = tid; i < valid; i += 256) sum += __expf(s[i] - mx);
    red[tid] = sum; __syncthreads();
    for (int w = 128; w > 0; w >>= 1) {
        if (tid < w) red[tid] += red[tid + w];
        __syncthreads();
    }
    const float inv = 1.0f / red[0];

    for (int i = tid; i < valid; i += 256) p[i] = (bf16)(__expf(s[i] - mx) * inv);
    for (int i = valid + tid; i < T; i += 256) p[i] = (bf16)0.0f;
}

// ---------------------------------------------------------------------------
// out = LayerNorm(a + res) * g + b ; writes f32 (and optional bf16 copy)
// D = 1024, 256 threads, 4 elements/thread held in registers.
// ---------------------------------------------------------------------------
__global__ __launch_bounds__(256)
void ln_residual(const float* __restrict__ a, const float* __restrict__ res,
                 const float* __restrict__ g, const float* __restrict__ b,
                 float* __restrict__ outf, bf16* __restrict__ outb, int D) {
    const int row = blockIdx.x;
    const size_t base = (size_t)row * D;
    const int tid = threadIdx.x;
    float v[4];
#pragma unroll
    for (int e = 0; e < 4; ++e) {
        int i = tid + e * 256;
        v[e] = a[base + i] + res[base + i];
    }
    __shared__ float red[256];
    red[tid] = v[0] + v[1] + v[2] + v[3];
    __syncthreads();
    for (int w = 128; w > 0; w >>= 1) {
        if (tid < w) red[tid] += red[tid + w];
        __syncthreads();
    }
    const float mu = red[0] / (float)D; __syncthreads();

    float q = 0.f;
#pragma unroll
    for (int e = 0; e < 4; ++e) { float d = v[e] - mu; q += d * d; }
    red[tid] = q; __syncthreads();
    for (int w = 128; w > 0; w >>= 1) {
        if (tid < w) red[tid] += red[tid + w];
        __syncthreads();
    }
    const float rinv = rsqrtf(red[0] / (float)D + LN_EPS);
#pragma unroll
    for (int e = 0; e < 4; ++e) {
        int i = tid + e * 256;
        float y = (v[e] - mu) * rinv * g[i] + b[i];
        outf[base + i] = y;
        if (outb) outb[base + i] = (bf16)y;
    }
}

// ---------------------------------------------------------------------------
extern "C" void kernel_launch(void* const* d_in, const int* in_sizes, int n_in,
                              void* d_out, int out_size, void* d_ws, size_t ws_size,
                              hipStream_t stream) {
    const float* x     = (const float*)d_in[0];
    const float* wq    = (const float*)d_in[1];
    const float* wk    = (const float*)d_in[2];
    const float* wv    = (const float*)d_in[3];
    const float* wu    = (const float*)d_in[4];
    const float* bu    = (const float*)d_in[5];
    const float* ln0_g = (const float*)d_in[6];
    const float* ln0_b = (const float*)d_in[7];
    const float* w1    = (const float*)d_in[8];
    const float* b1    = (const float*)d_in[9];
    const float* w2    = (const float*)d_in[10];
    const float* b2    = (const float*)d_in[11];
    const float* ln1_g = (const float*)d_in[12];
    const float* ln1_b = (const float*)d_in[13];
    float* out = (float*)d_out;

    char* ws = (char*)d_ws;
    size_t off = 0;
    auto alloc = [&](size_t bytes) -> void* {
        void* p = ws + off;
        off += (bytes + 255) & ~(size_t)255;
        return p;
    };

    bf16* xb  = (bf16*)alloc((size_t)BT * EMB * 2);
    bf16* wqb = (bf16*)alloc((size_t)EMB * HD * 2);
    bf16* wkb = (bf16*)alloc((size_t)EMB * HD * 2);
    bf16* wvb = (bf16*)alloc((size_t)EMB * HD * 2);
    bf16* wub = (bf16*)alloc((size_t)HD * EMB * 2);
    bf16* w1b = (bf16*)alloc((size_t)EMB * HID * 2);
    bf16* w2b = (bf16*)alloc((size_t)HID * EMB * 2);
    bf16* Qb  = (bf16*)alloc((size_t)BT * HD * 2);     // [b,t,h,d]
    bf16* Kb  = (bf16*)alloc((size_t)BT * HD * 2);
    bf16* Vb  = (bf16*)alloc((size_t)BT * HD * 2);
    float* Sf = (float*)alloc((size_t)BATCH * HEADS * SEQ * SEQ * 4);
    bf16* Pb  = (bf16*)alloc((size_t)BATCH * HEADS * SEQ * SEQ * 2);
    bf16* Yb  = (bf16*)alloc((size_t)BT * HD * 2);     // [b,t,h,d]
    float* AO = (float*)alloc((size_t)BT * EMB * 4);
    float* X1f= (float*)alloc((size_t)BT * EMB * 4);
    bf16* X1b = (bf16*)alloc((size_t)BT * EMB * 2);
    bf16* H1  = (bf16*)alloc((size_t)BT * HID * 2);
    float* WO = (float*)alloc((size_t)BT * EMB * 4);

    const float scale = 0.17677669529663687f;          // EMB^-0.25

    auto cvt = [&](const float* in, bf16* o, size_t n) {
        unsigned int blocks = (unsigned int)((n + 256 * 8 - 1) / (256 * 8));
        cvt_f32_bf16<<<blocks, 256, 0, stream>>>(in, o, n);
    };
    cvt(x,  xb,  (size_t)BT * EMB);
    cvt(wq, wqb, (size_t)EMB * HD);
    cvt(wk, wkb, (size_t)EMB * HD);
    cvt(wv, wvb, (size_t)EMB * HD);
    cvt(wu, wub, (size_t)HD * EMB);
    cvt(w1, w1b, (size_t)EMB * HID);
    cvt(w2, w2b, (size_t)HID * EMB);

    // ---- QKV projections: [BT,EMB] @ [EMB,HD] -> [b,t,h,d], q/k scaled ----
    dim3 gqkv(HD / BN, BT / BM, 1);
    gemm_wmma<false, false, bf16><<<gqkv, GT, 0, stream>>>(
        xb, wqb, Qb, BT, HD, EMB, EMB, HD, HD,
        0, 0, 0, 0, 0, 0, 1, nullptr, scale, 0);
    gemm_wmma<false, false, bf16><<<gqkv, GT, 0, stream>>>(
        xb, wkb, Kb, BT, HD, EMB, EMB, HD, HD,
        0, 0, 0, 0, 0, 0, 1, nullptr, scale, 0);
    gemm_wmma<false, false, bf16><<<gqkv, GT, 0, stream>>>(
        xb, wvb, Vb, BT, HD, EMB, EMB, HD, HD,
        0, 0, 0, 0, 0, 0, 1, nullptr, 1.0f, 0);

    // ---- scores: S[b,h] = Q[b,:,h,:] @ K[b,:,h,:]^T  (causal block skip) ----
    dim3 gsc(SEQ / BN, SEQ / BM, BATCH * HEADS);
    gemm_wmma<true, false, float><<<gsc, GT, 0, stream>>>(
        Qb, Kb, Sf, SEQ, SEQ, EMB, HD, HD, SEQ,
        (size_t)SEQ * HD, (size_t)EMB,          // A: b-stride, h-stride
        (size_t)SEQ * HD, (size_t)EMB,          // B: b-stride, h-stride
        (size_t)HEADS * SEQ * SEQ, (size_t)SEQ * SEQ,
        HEADS, nullptr, 1.0f, 1);

    // ---- causal softmax -> bf16 probabilities ----
    softmax_causal<<<BATCH * HEADS * SEQ, 256, 0, stream>>>(Sf, Pb, SEQ);

    // ---- Y[b,:,h,:] = P[b,h] @ V[b,:,h,:] ----
    dim3 gpv(EMB / BN, SEQ / BM, BATCH * HEADS);
    gemm_wmma<false, false, bf16><<<gpv, GT, 0, stream>>>(
        Pb, Vb, Yb, SEQ, EMB, SEQ, SEQ, HD, HD,
        (size_t)HEADS * SEQ * SEQ, (size_t)SEQ * SEQ,
        (size_t)SEQ * HD, (size_t)EMB,
        (size_t)SEQ * HD, (size_t)EMB,
        HEADS, nullptr, 1.0f, 0);

    // ---- attn_out = Y[BT,HD] @ wu[HD,EMB] + bu ----
    dim3 gwu(EMB / BN, BT / BM, 1);
    gemm_wmma<false, false, float><<<gwu, GT, 0, stream>>>(
        Yb, wub, AO, BT, EMB, HD, HD, EMB, EMB,
        0, 0, 0, 0, 0, 0, 1, bu, 1.0f, 0);

    // ---- x1 = LN(attn_out + x) ----
    ln_residual<<<BT, 256, 0, stream>>>(AO, x, ln0_g, ln0_b, X1f, X1b, EMB);

    // ---- FFN: H1 = relu(x1 @ w1 + b1) ----
    dim3 gf1(HID / BN, BT / BM, 1);
    gemm_wmma<false, true, bf16><<<gf1, GT, 0, stream>>>(
        X1b, w1b, H1, BT, HID, EMB, EMB, HID, HID,
        0, 0, 0, 0, 0, 0, 1, b1, 1.0f, 0);

    // ---- WO = H1 @ w2 + b2 ----
    dim3 gf2(EMB / BN, BT / BM, 1);
    gemm_wmma<false, false, float><<<gf2, GT, 0, stream>>>(
        H1, w2b, WO, BT, EMB, HID, HID, EMB, EMB,
        0, 0, 0, 0, 0, 0, 1, b2, 1.0f, 0);

    // ---- out = LN(WO + x1) ----
    ln_residual<<<BT, 256, 0, stream>>>(WO, X1f, ln1_g, ln1_b, out, nullptr, EMB);
}
